// PhaMPN_33741263078268
// MI455X (gfx1250) — matrix operand order repaired and checked by reading, not compiled, not run on hardware
//
#include <hip/hip_runtime.h>

// ---------------------------------------------------------------------------
// MPNN (chemprop-style) for gfx1250 / MI455X.
// message kept in f16 (fits L2: 38.4MB), all GEMMs via v_wmma_f32_16x16x32_f16
// with f32 accumulate + fast-sigmoid (v_exp + v_rcp) epilogue.
// Round 2: preload all WMMA fragments (overlap loads/WMMA, single wait),
//          replace precise-divide sigmoid with v_rcp path, clause gathers.
// ---------------------------------------------------------------------------

typedef __attribute__((ext_vector_type(16))) _Float16 v16h;
typedef __attribute__((ext_vector_type(8)))  _Float16 v8h;
typedef __attribute__((ext_vector_type(8)))  float    v8f;

#define HID 128
#define NNB 8
#define FFE 41   // FF + EF
#define FFO 40   // FF
#define NMOL 500

// sigmoid via hardware transcendentals: v_exp_f32 + v_rcp_f32 (no div_fixup)
__device__ __forceinline__ float sigf(float x) {
    return __builtin_amdgcn_rcpf(1.0f + __expf(-x));
}

// ---------------------------------------------------------------------------
// Prep: W_h (128x128, [k][n]) -> WhT f16 [n][k]; W_o rows 40..167 -> WoHiT [n][k]
// ---------------------------------------------------------------------------
__global__ __launch_bounds__(256) void prep_weights_kernel(
    const float* __restrict__ W_h, const float* __restrict__ W_o,
    _Float16* __restrict__ WhT, _Float16* __restrict__ WoHiT)
{
    int idx = blockIdx.x * blockDim.x + threadIdx.x;      // 0 .. 128*128-1
    if (idx >= HID * HID) return;
    int k = idx & (HID - 1);
    int n = idx >> 7;
    WhT  [n * HID + k] = (_Float16)W_h[(size_t)k * HID + n];
    WoHiT[n * HID + k] = (_Float16)W_o[(size_t)(FFO + k) * HID + n];
}

// ---------------------------------------------------------------------------
// Init: binput = fedges @ W_i ; msg = sigmoid(binput) in f16
// ---------------------------------------------------------------------------
__global__ __launch_bounds__(256) void init_kernel(
    const float* __restrict__ fedges, const float* __restrict__ W_i,
    float* __restrict__ binput, _Float16* __restrict__ msg, int E)
{
    int idx = blockIdx.x * blockDim.x + threadIdx.x;
    if (idx >= E * HID) return;
    int e = idx >> 7;
    int h = idx & (HID - 1);
    const float* fr = fedges + (size_t)e * FFE;
    float acc = 0.f;
#pragma unroll
    for (int k = 0; k < FFE; ++k) acc += fr[k] * W_i[(size_t)k * HID + h];
    binput[idx] = acc;
    msg[idx] = (_Float16)sigf(acc);
}

// ---------------------------------------------------------------------------
// Partial: partial = features @ W_o[0:40] + b_o   (f32, node side)
// ---------------------------------------------------------------------------
__global__ __launch_bounds__(256) void partial_kernel(
    const float* __restrict__ features, const float* __restrict__ W_o,
    const float* __restrict__ b_o, float* __restrict__ partial, int N)
{
    int idx = blockIdx.x * blockDim.x + threadIdx.x;
    if (idx >= N * HID) return;
    int n = idx >> 7;
    int h = idx & (HID - 1);
    const float* fr = features + (size_t)n * FFO;
    float acc = b_o[h];
#pragma unroll
    for (int k = 0; k < FFO; ++k) acc += fr[k] * W_o[(size_t)k * HID + h];
    partial[idx] = acc;
}

// ---------------------------------------------------------------------------
// Fused step: A-tile = gather-sum_{j<8} msg_in[graph[e,j]]  (16 rows x 128, LDS f16)
//             C = A @ WT  via 4x v_wmma_f32_16x16x32_f16 per 16x16 N-tile
//             if SIG16: msg_out = f16(sigmoid(bias + C))   (edge message step)
//             else    : out_f32 = sigmoid(bias + C)        (final node layer)
// One block = 256 threads = 8 wave32; block owns 16 rows, wave w owns n-tile w.
// ---------------------------------------------------------------------------
template <bool SIG16>
__global__ __launch_bounds__(256) void fused_gemm_kernel(
    const _Float16* __restrict__ msg_in,   // [E, 128] f16 messages (gather source)
    const int*      __restrict__ graph,    // [M, 8] neighbor edge indices
    const float*    __restrict__ bias,     // [M, 128] f32 (binput or partial)
    const _Float16* __restrict__ WT,       // [128, 128] f16, [n][k] (pre-transposed)
    _Float16*       __restrict__ out_h,    // [M, 128] f16 (when SIG16)
    float*          __restrict__ out_f)    // [M, 128] f32 (when !SIG16)
{
    __shared__ _Float16 Atile[16][HID + 8];          // 272B row stride, 16B aligned

    const int r0 = blockIdx.x * 16;                  // first row of this tile
    const int t  = threadIdx.x;

    // ---- phase 1: gather-sum 16x128 A-tile into LDS (f32 accumulate -> f16) ----
    {
        const int r = t >> 4;                        // 0..15 row in tile
        const int c = (t & 15) * 8;                  // 8 f16 columns per thread
        const int* g = graph + (size_t)(r0 + r) * NNB;
        int idxs[NNB];
        *(int4*)&idxs[0] = *(const int4*)(g);        // clause the index loads
        *(int4*)&idxs[4] = *(const int4*)(g + 4);
        v8h mv[NNB];                                 // 8 independent b128 gathers
#pragma unroll
        for (int j = 0; j < NNB; ++j)
            mv[j] = *(const v8h*)(msg_in + (size_t)idxs[j] * HID + c);
        float acc[8];
#pragma unroll
        for (int i = 0; i < 8; ++i) acc[i] = 0.f;
#pragma unroll
        for (int j = 0; j < NNB; ++j)
#pragma unroll
            for (int i = 0; i < 8; ++i) acc[i] += (float)mv[j][i];
#pragma unroll
        for (int i = 0; i < 8; ++i) Atile[r][c + i] = (_Float16)acc[i];
    }
    __syncthreads();

    // ---- phase 2: each wave computes one 16x16 output tile, K=128 = 4 x K32 ----
    const int wave = t >> 5;                         // 0..7 -> n-tile
    const int lane = t & 31;
    const int hi   = lane >> 4;                      // half-wave select
    const int ln   = lane & 15;
    const int n0   = wave * 16;

    // preload ALL fragments first so the 4 WMMAs issue back-to-back after one wait
    v16h afrag[4], bfrag[4];
#pragma unroll
    for (int kt = 0; kt < 4; ++kt) {
        const int k0 = kt * 32;
        // A fragment: row=ln; lanes0-15 K={k0..k0+15},
        //                     lanes16-31 K={k0+16..k0+31}
        const v8h a0 = *(const v8h*)&Atile[ln][k0 + hi * 16];
        const v8h a1 = *(const v8h*)&Atile[ln][k0 + hi * 16 + 8];
#pragma unroll
        for (int i = 0; i < 8; ++i) { afrag[kt][i] = a0[i]; afrag[kt][i + 8] = a1[i]; }
        // B fragment: col = n0+ln, 16 consecutive K per half-wave (WT is [n][k])
        bfrag[kt] = *(const v16h*)(WT + (size_t)(n0 + ln) * HID + k0 + hi * 16);
    }

    v8f cacc = {};
#pragma unroll
    for (int kt = 0; kt < 4; ++kt)
        cacc = __builtin_amdgcn_wmma_f32_16x16x32_f16(
            /*neg_a=*/false, afrag[kt], /*neg_b=*/false, bfrag[kt],
            /*c_mod=*/(short)0, cacc, /*reuse_a=*/false, /*reuse_b=*/false);

    // ---- epilogue: C layout: VGPR i -> M = i + hi*8, N = n0 + ln ----
    // hoist the 8 strided bias loads into one clause
    const size_t colo = (size_t)(n0 + ln);
    float bvals[8];
#pragma unroll
    for (int i = 0; i < 8; ++i)
        bvals[i] = bias[(size_t)(r0 + i + hi * 8) * HID + colo];
#pragma unroll
    for (int i = 0; i < 8; ++i) {
        const size_t off = (size_t)(r0 + i + hi * 8) * HID + colo;
        const float v = sigf(bvals[i] + cacc[i]);
        if (SIG16) out_h[off] = (_Float16)v;
        else       out_f[off] = v;
    }
}

// ---------------------------------------------------------------------------
// Segment mean over molecules: out[b][h] = mean(hidden[start..start+len][h])
// ---------------------------------------------------------------------------
__global__ __launch_bounds__(128) void reduce_kernel(
    const float* __restrict__ hidden, const int* __restrict__ scope,
    float* __restrict__ out)
{
    const int b = blockIdx.x;
    const int h = threadIdx.x;
    const int start = scope[b * 2 + 0];
    const int len   = scope[b * 2 + 1];
    float s = 0.f;
    for (int i = 0; i < len; ++i)
        s += hidden[(size_t)(start + i) * HID + h];
    out[(size_t)b * HID + h] = s / (float)len;
}

// ---------------------------------------------------------------------------
extern "C" void kernel_launch(void* const* d_in, const int* in_sizes, int n_in,
                              void* d_out, int out_size, void* d_ws, size_t ws_size,
                              hipStream_t stream)
{
    const float* features = (const float*)d_in[0];   // [N, 40]
    const float* fedges   = (const float*)d_in[1];   // [E, 41]
    const int*   agraph   = (const int*)  d_in[2];   // [N, 8]
    const int*   egraph   = (const int*)  d_in[3];   // [E, 8]
    const int*   scope    = (const int*)  d_in[4];   // [B, 2]
    const float* W_i      = (const float*)d_in[5];   // [41, 128]
    const float* W_h      = (const float*)d_in[6];   // [128, 128]
    const float* W_o      = (const float*)d_in[7];   // [168, 128]
    const float* b_o      = (const float*)d_in[8];   // [128]
    float* out = (float*)d_out;                      // [500, 128]

    const int N = in_sizes[0] / FFO;                 // 50000 (multiple of 16)
    const int E = in_sizes[1] / FFE;                 // 150000 (multiple of 16)

    // --- carve workspace (≈205 MB total) ---
    char* ws = (char*)d_ws;
    auto carve = [&](size_t bytes) {
        char* p = ws;
        ws += (bytes + 255) & ~(size_t)255;
        return p;
    };
    float*     binput = (float*)    carve((size_t)E * HID * sizeof(float));
    _Float16*  msgA   = (_Float16*) carve((size_t)E * HID * sizeof(_Float16));
    _Float16*  msgB   = (_Float16*) carve((size_t)E * HID * sizeof(_Float16));
    float*     part   = (float*)    carve((size_t)N * HID * sizeof(float));
    float*     hidden = (float*)    carve((size_t)N * HID * sizeof(float));
    _Float16*  WhT    = (_Float16*) carve((size_t)HID * HID * sizeof(_Float16));
    _Float16*  WoHiT  = (_Float16*) carve((size_t)HID * HID * sizeof(_Float16));

    // --- prep: f16 transposed weights ---
    prep_weights_kernel<<<(HID * HID + 255) / 256, 256, 0, stream>>>(W_h, W_o, WhT, WoHiT);

    // --- init: binput + message0 ---
    init_kernel<<<((size_t)E * HID + 255) / 256, 256, 0, stream>>>(fedges, W_i, binput, msgA, E);

    // --- node-side partial: features @ W_o[:40] + b_o ---
    partial_kernel<<<((size_t)N * HID + 255) / 256, 256, 0, stream>>>(features, W_o, b_o, part, N);

    // --- 5 message-passing steps (DEPTH-1), ping-pong msgA/msgB ---
    _Float16* cur = msgA;
    _Float16* nxt = msgB;
    for (int it = 0; it < 5; ++it) {
        fused_gemm_kernel<true><<<E / 16, 256, 0, stream>>>(
            cur, egraph, binput, WhT, nxt, nullptr);
        _Float16* tmp = cur; cur = nxt; nxt = tmp;
    }

    // --- final node layer: gather over agraph + WMMA with W_o[40:,:] + sigmoid ---
    fused_gemm_kernel<false><<<N / 16, 256, 0, stream>>>(
        cur, agraph, part, WoHiT, nullptr, hidden);

    // --- segment mean ---
    reduce_kernel<<<NMOL, HID, 0, stream>>>(hidden, scope, out);
}
